// MultiheadAttention_46935402611087
// MI455X (gfx1250) — matrix operand order repaired and checked by reading, not compiled, not run on hardware
//
#include <hip/hip_runtime.h>
#include <hip/hip_bf16.h>

// ---------------------------------------------------------------------------
// MI455X (gfx1250) fused xPos multihead attention.
// Wave32; WMMA f32_16x16x32_f16 for all GEMMs; score tile resident in the
// 320KB WGP LDS (w written to HBM exactly once); K tiles staged via the
// CDNA5 Tensor Data Mover (tensor_load_to_lds + s_wait_tensorcnt).
// ---------------------------------------------------------------------------

#define B_SZ 4
#define T_SZ 1024
#define E_SZ 1024
#define H_SZ 16
#define D_SZ 64
#define BH_SZ (B_SZ * H_SZ)      // 64 batch-heads
#define M_ROWS (B_SZ * T_SZ)     // 4096

typedef __attribute__((ext_vector_type(16))) _Float16 v16h;
typedef __attribute__((ext_vector_type(8)))  _Float16 v8h;
typedef __attribute__((ext_vector_type(8)))  float    v8f;
typedef __attribute__((ext_vector_type(4)))  unsigned int u32x4;
typedef __attribute__((ext_vector_type(8)))  unsigned int u32x8;

// ===========================================================================
// C = alpha * (A[M,K] @ W[N,K]^T + bias)  with WMMA f16 math, f32 accumulate.
// MODE 0: fp32 row-major [M,N]
// MODE 1: fp32 head-split [B*H, T, D]
// MODE 2: f16 transposed head-split [B*H, D, T]  (for V)
// ===========================================================================
#define LDS_AW 40   // padded f16 row stride for a 32-wide K slab

template <int MODE>
__global__ __launch_bounds__(128) void gemm_xwt_kernel(
    const float* __restrict__ A, const float* __restrict__ W,
    const float* __restrict__ bias, float alpha,
    float* __restrict__ outF, _Float16* __restrict__ outH,
    int M, int N, int K)
{
  __shared__ _Float16 sA[64 * LDS_AW];
  __shared__ _Float16 sW[64 * LDS_AW];

  const int tid  = threadIdx.x;
  const int lane = tid & 31;
  const int wave = tid >> 5;          // 4 waves, 2x2 over the 64x64 tile
  const int wy = wave >> 1, wx = wave & 1;
  const int m0 = blockIdx.x * 64;
  const int n0 = blockIdx.y * 64;
  const int lh = lane & 15, lhi = lane >> 4;

  // staging map: thread -> (row = tid/4, col8 = (tid&3)*8), two row passes
  const int sr = tid >> 2;            // 0..31
  const int sc = (tid & 3) * 8;       // 0,8,16,24

  v8f acc[2][2] = {};

  for (int k0 = 0; k0 < K; k0 += 32) {
    if (k0 + 32 < K) {  // CDNA5 global_prefetch_b8 for next slabs
      __builtin_prefetch(&A[(size_t)(m0 + (tid >> 1)) * K + k0 + 32], 0, 1);
      __builtin_prefetch(&W[(size_t)(n0 + (tid >> 1)) * K + k0 + 32], 0, 1);
    }
    __syncthreads();
    // Stage 64x32 fp32 slabs of A and W into LDS as f16: vectorized b128
    // global loads + packed b128 LDS stores (no per-element waits).
#pragma unroll
    for (int hp = 0; hp < 2; ++hp) {
      int r = sr + 32 * hp;
      const float4* ap = (const float4*)&A[(size_t)(m0 + r) * K + k0 + sc];
      const float4* wp = (const float4*)&W[(size_t)(n0 + r) * K + k0 + sc];
      float4 a0 = ap[0], a1 = ap[1];
      float4 w0 = wp[0], w1 = wp[1];
      v8h ah, wh;
      ah[0] = (_Float16)a0.x; ah[1] = (_Float16)a0.y; ah[2] = (_Float16)a0.z; ah[3] = (_Float16)a0.w;
      ah[4] = (_Float16)a1.x; ah[5] = (_Float16)a1.y; ah[6] = (_Float16)a1.z; ah[7] = (_Float16)a1.w;
      wh[0] = (_Float16)w0.x; wh[1] = (_Float16)w0.y; wh[2] = (_Float16)w0.z; wh[3] = (_Float16)w0.w;
      wh[4] = (_Float16)w1.x; wh[5] = (_Float16)w1.y; wh[6] = (_Float16)w1.z; wh[7] = (_Float16)w1.w;
      *(v8h*)&sA[r * LDS_AW + sc] = ah;
      *(v8h*)&sW[r * LDS_AW + sc] = wh;
    }
    __syncthreads();

    // A fragments (16x32, ISA 16-bit A layout), B fragments (32x16).
    v16h af[2], bf[2];
#pragma unroll
    for (int im = 0; im < 2; ++im) {
      const _Float16* base = &sA[(32 * wy + 16 * im + lh) * LDS_AW];
      v8h lo = *(const v8h*)(base + (lhi ? 8 : 0));
      v8h hi = *(const v8h*)(base + 16 + (lhi ? 8 : 0));
#pragma unroll
      for (int i = 0; i < 8; ++i) { af[im][i] = lo[i]; af[im][8 + i] = hi[i]; }
    }
#pragma unroll
    for (int jn = 0; jn < 2; ++jn) {
      const _Float16* base = &sW[(32 * wx + 16 * jn + lh) * LDS_AW] + (lhi ? 8 : 0);
      v8h lo = *(const v8h*)(base);
      v8h hi = *(const v8h*)(base + 16);
#pragma unroll
      for (int i = 0; i < 8; ++i) { bf[jn][i] = lo[i]; bf[jn][8 + i] = hi[i]; }
    }
#pragma unroll
    for (int im = 0; im < 2; ++im)
#pragma unroll
      for (int jn = 0; jn < 2; ++jn)
        acc[im][jn] = __builtin_amdgcn_wmma_f32_16x16x32_f16(
            false, af[im], false, bf[jn], (short)0, acc[im][jn], false, false);
  }

  // Epilogue: bias + alpha + layout (straight-line per MODE).
#pragma unroll
  for (int im = 0; im < 2; ++im) {
#pragma unroll
    for (int jn = 0; jn < 2; ++jn) {
      int col = n0 + 32 * wx + 16 * jn + lh;
      float bv = bias[col];
#pragma unroll
      for (int i = 0; i < 8; ++i) {
        int row = m0 + 32 * wy + 16 * im + i + 8 * lhi;
        float v = alpha * (acc[im][jn][i] + bv);
        if (MODE == 0) {
          outF[(size_t)row * N + col] = v;
        } else {
          int b = row >> 10, t = row & (T_SZ - 1);
          int h = col >> 6,  d = col & (D_SZ - 1);
          if (MODE == 1)
            outF[((size_t)(b * H_SZ + h) * T_SZ + t) * D_SZ + d] = v;
          else  // MODE 2: f16, [BH, D, T] so V WMMA B-frags are contiguous
            outH[((size_t)(b * H_SZ + h) * D_SZ + d) * T_SZ + t] = (_Float16)v;
        }
      }
    }
  }
}

// ===========================================================================
// xPos rotary scaling:  out = x*cos*s + rot(x)*sin*s  (s or 1/s)
// ===========================================================================
__global__ __launch_bounds__(256) void xpos_kernel(
    const float* __restrict__ xin, _Float16* __restrict__ xout, int downscale)
{
  int idx = blockIdx.x * 256 + threadIdx.x;          // one even/odd pair each
  if (idx >= BH_SZ * T_SZ * (D_SZ / 2)) return;
  int j  = idx & 31;
  int t  = (idx >> 5) & (T_SZ - 1);
  int bh = idx >> 15;

  float sv    = (2.0f * j + 0.4f * D_SZ) / (1.4f * D_SZ);
  float power = ((float)t - (float)(T_SZ / 2)) / 512.0f;   // SCALE_BASE
  float scale = powf(sv, power);
  if (downscale) scale = 1.0f / scale;
  float inv_freq = powf(10000.0f, -(float)j / 32.0f);
  float ang = (float)t * inv_freq;
  float s = __sinf(ang) * scale, c = __cosf(ang) * scale;

  size_t base = ((size_t)bh * T_SZ + t) * D_SZ + 2 * j;
  float x1 = xin[base], x2 = xin[base + 1];
  xout[base]     = (_Float16)(x1 * c - x2 * s);
  xout[base + 1] = (_Float16)(x2 * c + x1 * s);
}

// ===========================================================================
// Fused attention: one workgroup = one (bh, 64-q-row tile).
// S tile 64x1024 fp32 in LDS (stride 1025 -> conflict-free).  K tiles staged
// into LDS by the Tensor Data Mover (1-D 8KB tile, LDS padding 16B per 128B
// so rows land at a 72-half stride: bank-conflict-free AND 16B aligned).
// ===========================================================================
#define S_STRIDE  1025
#define KT_STRIDE 72                       // halves per 64-half row after pad
#define TDIM0     (1u << 20)               // generous 1-D tensor extent

__global__ __launch_bounds__(128) void attn_kernel(
    const _Float16* __restrict__ qh,   // [BH,T,D] f16 (xpos'd, scaled)
    const _Float16* __restrict__ kh,   // [BH,T,D] f16 (xpos'd)
    const _Float16* __restrict__ vT,   // [BH,D,T] f16 (transposed)
    const float* __restrict__ rel,     // [H,T,T]
    float* __restrict__ wout,          // [H,B,T,T]
    float* __restrict__ attn)          // [B,T,E]
{
  extern __shared__ float smem[];
  float* S    = smem;                          // [64][S_STRIDE]
  float* red  = smem + 64 * S_STRIDE;          // [64][2]
  float* rowv = red + 128;                     // [64]
  _Float16* kTile = (_Float16*)(rowv + 64);    // [64][KT_STRIDE] (64B aligned)

  const int tid  = threadIdx.x;
  const int lane = tid & 31, wave = tid >> 5;
  const int lh = lane & 15,  lhi = lane >> 4;
  const int r0 = blockIdx.x * 64;
  const int bh = blockIdx.y;
  const int b = bh >> 4, h = bh & 15;

  const _Float16* qbase = qh + (size_t)bh * T_SZ * D_SZ;
  const _Float16* kbase = kh + (size_t)bh * T_SZ * D_SZ;
  const _Float16* vbase = vT + (size_t)bh * D_SZ * T_SZ;
  const float* relbase  = rel + (size_t)h * T_SZ * T_SZ;
  float* wbase = wout + ((size_t)h * B_SZ + b) * T_SZ * T_SZ;

  // Q fragments for this wave (rows r0+16*wave..+15, K=64 -> 2 frags),
  // register-resident for all 16 key tiles.
  v16h qf[2];
  {
    const _Float16* p = qbase + (size_t)(r0 + 16 * wave + lh) * D_SZ;
#pragma unroll
    for (int kk = 0; kk < 2; ++kk) {
      v8h lo = *(const v8h*)(p + kk * 32 + (lhi ? 8 : 0));
      v8h hi = *(const v8h*)(p + kk * 32 + 16 + (lhi ? 8 : 0));
#pragma unroll
      for (int i = 0; i < 8; ++i) { qf[kk][i] = lo[i]; qf[kk][8 + i] = hi[i]; }
    }
  }

  // ---- Phase 1: S = q @ k^T + rel_pos, into LDS -------------------------
  for (int s0 = 0; s0 < T_SZ; s0 += 64) {
    __syncthreads();   // everyone done with previous kTile
    if (wave == 0) {
      // TDM: DMA the contiguous 64x64 f16 K tile into LDS with row padding.
      unsigned long long ga =
          (unsigned long long)(size_t)(kbase + (size_t)s0 * D_SZ);
      unsigned ldsoff = (unsigned)(size_t)kTile;   // flat low 32b = LDS offset
      u32x4 g0; u32x8 g1; u32x4 gz;
      g0[0] = 1u;                                   // count=1, user mode
      g0[1] = ldsoff;                               // lds_addr
      g0[2] = (unsigned)ga;                         // global_addr[31:0]
      g0[3] = ((unsigned)(ga >> 32) & 0x01FFFFFFu)  // global_addr[56:32]
              | 0x80000000u;                        // type=2 (bits 127:126=10b)
      g1[0] = (1u << 16)       // data_size = 2B
            | (1u << 20)       // pad_enable
            | (4u << 22)       // pad_interval: every 32 DWORDs (128B row)
            | (3u << 25);      // pad_amount: 4 DWORDs (16B) -> 72-half stride
      g1[1] = (TDIM0 & 0xFFFFu) << 16;              // tensor_dim0 lo
      g1[2] = (TDIM0 >> 16) | (1u << 16);           // tensor_dim0 hi, dim1=1
      g1[3] = (4096u << 16);                        // tile_dim0 = 4096 elems
      g1[4] = 0u;                                   // tile_dim1/2 unused (1-D)
      g1[5] = TDIM0;                                // tensor_dim0_stride lo
      g1[6] = 0u; g1[7] = 0u;
      gz[0] = 0u; gz[1] = 0u; gz[2] = 0u; gz[3] = 0u;
      asm volatile("tensor_load_to_lds %0, %1, %2, %3"
                   :: "s"(g0), "s"(g1), "s"(gz), "s"(gz) : "memory");
      __builtin_amdgcn_s_wait_tensorcnt(0);
    }
    __syncthreads();   // kTile published

    v8f c[4] = {};
#pragma unroll
    for (int jn = 0; jn < 4; ++jn) {
      const _Float16* kp = kTile + (16 * jn + lh) * KT_STRIDE + (lhi ? 8 : 0);
#pragma unroll
      for (int kk = 0; kk < 2; ++kk) {
        v8h lo = *(const v8h*)(kp + kk * 32);
        v8h hi = *(const v8h*)(kp + kk * 32 + 16);
        v16h bfr;
#pragma unroll
        for (int i = 0; i < 8; ++i) { bfr[i] = lo[i]; bfr[8 + i] = hi[i]; }
        c[jn] = __builtin_amdgcn_wmma_f32_16x16x32_f16(
            false, qf[kk], false, bfr, (short)0, c[jn], false, false);
      }
    }
#pragma unroll
    for (int jn = 0; jn < 4; ++jn) {
      int scol = s0 + 16 * jn + lh;
#pragma unroll
      for (int i = 0; i < 8; ++i) {
        int row = 16 * wave + i + 8 * lhi;   // local q-row (wave-private)
        float v = c[jn][i] + relbase[(size_t)(r0 + row) * T_SZ + scol];
        S[row * S_STRIDE + scol] = v;
      }
    }
  }
  __syncthreads();

  // ---- Phase 2: exact softmax over LDS-resident rows --------------------
  {
    int r = tid & 63, half = tid >> 6;
    float* Sr = S + r * S_STRIDE + half * 512;
    float m = -1e30f;
    for (int c2 = 0; c2 < 512; ++c2) m = fmaxf(m, Sr[c2]);
    red[r * 2 + half] = m;
    __syncthreads();
    if (tid < 64) rowv[tid] = fmaxf(red[tid * 2], red[tid * 2 + 1]);
    __syncthreads();
    float rm = rowv[r], sum = 0.0f;
    for (int c2 = 0; c2 < 512; ++c2) {
      float p = __expf(Sr[c2] - rm);
      Sr[c2] = p;
      sum += p;
    }
    red[r * 2 + half] = sum;
    __syncthreads();
    if (tid < 64) rowv[tid] = 1.0f / (red[tid * 2] + red[tid * 2 + 1]);
    __syncthreads();
  }
  // Normalize in LDS and stream w to HBM once, coalesced.
  for (int idx = tid; idx < 64 * T_SZ; idx += 128) {
    int r = idx >> 10, cc = idx & (T_SZ - 1);
    float p = S[r * S_STRIDE + cc] * rowv[r];
    S[r * S_STRIDE + cc] = p;
    wbase[(size_t)(r0 + r) * T_SZ + cc] = p;
  }
  __syncthreads();

  // ---- Phase 3: O(64x64) = P(64x1024) @ V(1024x64), P from LDS ----------
  v8f o[4] = {};
  for (int k0 = 0; k0 < T_SZ; k0 += 32) {
    v16h af;
    {
      const float* pr = S + (16 * wave + lh) * S_STRIDE + k0;
#pragma unroll
      for (int i = 0; i < 8; ++i) {
        af[i]     = (_Float16)pr[(lhi ? 8 : 0) + i];
        af[8 + i] = (_Float16)pr[16 + (lhi ? 8 : 0) + i];
      }
    }
#pragma unroll
    for (int jn = 0; jn < 4; ++jn) {
      const _Float16* vp = vbase + (size_t)(16 * jn + lh) * T_SZ + k0 + (lhi ? 8 : 0);
      v8h lo = *(const v8h*)(vp);
      v8h hi = *(const v8h*)(vp + 16);
      v16h bfr;
#pragma unroll
      for (int i = 0; i < 8; ++i) { bfr[i] = lo[i]; bfr[8 + i] = hi[i]; }
      o[jn] = __builtin_amdgcn_wmma_f32_16x16x32_f16(
          false, af, false, bfr, (short)0, o[jn], false, false);
    }
  }
  // Merge heads: attn[b][t][h*D + d]
#pragma unroll
  for (int jn = 0; jn < 4; ++jn) {
    int col = 16 * jn + lh;
#pragma unroll
    for (int i = 0; i < 8; ++i) {
      int row = 16 * wave + i + 8 * lhi;
      attn[((size_t)b * T_SZ + (r0 + row)) * E_SZ + h * D_SZ + col] = o[jn][i];
    }
  }
}

// ===========================================================================
extern "C" void kernel_launch(void* const* d_in, const int* in_sizes, int n_in,
                              void* d_out, int out_size, void* d_ws, size_t ws_size,
                              hipStream_t stream) {
  (void)in_sizes; (void)n_in; (void)out_size; (void)ws_size;
  const float* query  = (const float*)d_in[0];
  const float* key_in = (const float*)d_in[1];
  const float* value  = (const float*)d_in[2];
  const float* rel    = (const float*)d_in[3];
  const float* Wq = (const float*)d_in[4];  const float* bq = (const float*)d_in[5];
  const float* Wk = (const float*)d_in[6];  const float* bk = (const float*)d_in[7];
  const float* Wv = (const float*)d_in[8];  const float* bv = (const float*)d_in[9];
  const float* Wo = (const float*)d_in[10]; const float* bo = (const float*)d_in[11];

  float* outO = (float*)d_out;                              // [B,T,E]
  float* outW = outO + (size_t)B_SZ * T_SZ * E_SZ;          // [H,B,T,T]

  // Workspace (56 MB): qF/kF fp32 staging, q/k/v f16; qF reused for attn.
  char* ws = (char*)d_ws;
  float*    qF = (float*)ws;                       // 16 MB [BH,T,D] fp32
  float*    kF = (float*)(ws + (16 << 20));        // 16 MB
  _Float16* qh = (_Float16*)(ws + (32 << 20));     //  8 MB [BH,T,D] f16
  _Float16* kh = (_Float16*)(ws + (40 << 20));     //  8 MB
  _Float16* vh = (_Float16*)(ws + (48 << 20));     //  8 MB [BH,D,T] f16
  float*  attn = qF;                               // reuse after xpos(q)

  dim3 ggrid(M_ROWS / 64, E_SZ / 64);
  const float qscale = 0.125f;  // D^-0.5, folded into Q projection epilogue

  gemm_xwt_kernel<1><<<ggrid, 128, 0, stream>>>(query,  Wq, bq, qscale, qF, nullptr, M_ROWS, E_SZ, E_SZ);
  gemm_xwt_kernel<1><<<ggrid, 128, 0, stream>>>(key_in, Wk, bk, 1.0f,   kF, nullptr, M_ROWS, E_SZ, E_SZ);
  gemm_xwt_kernel<2><<<ggrid, 128, 0, stream>>>(value,  Wv, bv, 1.0f,   nullptr, vh, M_ROWS, E_SZ, E_SZ);

  int pairs = BH_SZ * T_SZ * (D_SZ / 2);
  xpos_kernel<<<(pairs + 255) / 256, 256, 0, stream>>>(qF, qh, 0);
  xpos_kernel<<<(pairs + 255) / 256, 256, 0, stream>>>(kF, kh, 1);

  size_t smem = (size_t)(64 * S_STRIDE + 128 + 64) * sizeof(float)   // S + red
              + (size_t)64 * KT_STRIDE * sizeof(_Float16);           // kTile
  attn_kernel<<<dim3(T_SZ / 64, BH_SZ), 128, smem, stream>>>(qh, kh, vh, rel, outW, attn);

  gemm_xwt_kernel<0><<<ggrid, 128, 0, stream>>>(attn, Wo, bo, 1.0f, outO, nullptr, M_ROWS, E_SZ, E_SZ);
}
